// TokenizeBlock_58351425683841
// MI455X (gfx1250) — matrix operand order repaired and checked by reading, not compiled, not run on hardware
//
#include <hip/hip_runtime.h>
#include <math.h>

typedef __attribute__((ext_vector_type(16))) _Float16 v16h;
typedef __attribute__((ext_vector_type(2)))  _Float16 h2v;
typedef __attribute__((ext_vector_type(8)))  float    v8f;

__device__ __forceinline__ unsigned short f32_to_f16_bits(float f) {
    _Float16 h = (_Float16)f;                    // v_cvt_f16_f32, RTE
    return __builtin_bit_cast(unsigned short, h);
}

// Branch-free erf-form GELU: Abramowitz-Stegun 7.1.26 (|err| < 1.5e-7),
// hardware v_rcp_f32 + v_exp_f32, no EXEC-mask divergence.
__device__ __forceinline__ float gelu_exact(float v) {
    const float a  = fabsf(v) * 0.70710678118654752f;   // |v|/sqrt(2)
    const float t  = __builtin_amdgcn_rcpf(1.0f + 0.3275911f * a);
    const float p  = t * (0.254829592f +
                     t * (-0.284496736f +
                     t * (1.421413741f +
                     t * (-1.453152027f +
                     t * 1.061405429f))));
    const float er = 1.0f - p * __expf(-a * a);          // erf(|v|/sqrt2)
    return 0.5f * v * (1.0f + copysignf(er, v));
}

// -------------------------------------------------------------------------
// Kernel 1: LayerNorm (E=512) + f16 cast.  One wave32 per token.
// Flat h layout: token t, feature e -> h[t*512 + e]  (== reshape to [M,K]).
// -------------------------------------------------------------------------
__global__ __launch_bounds__(256) void ln_to_f16_kernel(
    const float* __restrict__ x, const float* __restrict__ gamma,
    const float* __restrict__ beta, unsigned short* __restrict__ h, int ntok)
{
    const int lane = threadIdx.x & 31;
    const int wave = threadIdx.x >> 5;
    const int tok  = blockIdx.x * 8 + wave;
    if (tok >= ntok) return;

    const float* xp = x + (size_t)tok * 512 + lane * 16;
    float v[16];
#pragma unroll
    for (int i = 0; i < 4; ++i) {
        float4 t = ((const float4*)xp)[i];
        v[4*i+0] = t.x; v[4*i+1] = t.y; v[4*i+2] = t.z; v[4*i+3] = t.w;
    }
    float s = 0.f, s2 = 0.f;
#pragma unroll
    for (int i = 0; i < 16; ++i) { s += v[i]; s2 += v[i] * v[i]; }
#pragma unroll
    for (int off = 16; off > 0; off >>= 1) {
        s  += __shfl_xor(s,  off, 32);
        s2 += __shfl_xor(s2, off, 32);
    }
    const float mean = s * (1.0f / 512.0f);
    const float inv  = rsqrtf(s2 * (1.0f / 512.0f) - mean * mean + 1e-6f);

    const float* gp = gamma + lane * 16;
    const float* bp = beta  + lane * 16;
    unsigned int packed[8];
#pragma unroll
    for (int i = 0; i < 8; ++i) {
        float lo = (v[2*i+0] - mean) * inv * gp[2*i+0] + bp[2*i+0];
        float hi = (v[2*i+1] - mean) * inv * gp[2*i+1] + bp[2*i+1];
        packed[i] = (unsigned)f32_to_f16_bits(lo) | ((unsigned)f32_to_f16_bits(hi) << 16);
    }
    uint4* outp = (uint4*)(h + (size_t)tok * 512 + lane * 16);
    outp[0] = make_uint4(packed[0], packed[1], packed[2], packed[3]);
    outp[1] = make_uint4(packed[4], packed[5], packed[6], packed[7]);
}

// -------------------------------------------------------------------------
// Kernel 2: W [K][N] f32  ->  Wt [N][K] f16  (LDS-tiled transpose, one-time,
// 2 MB result lives in L2 for the GEMM).
// -------------------------------------------------------------------------
__global__ __launch_bounds__(256) void w_transpose_f16_kernel(
    const float* __restrict__ W, unsigned short* __restrict__ Wt, int K, int N)
{
    __shared__ unsigned short tile[32][33];
    const int kt = blockIdx.x * 32;
    const int nt = blockIdx.y * 32;
    const int tx = threadIdx.x & 31;
    const int ty = threadIdx.x >> 5;          // 0..7
#pragma unroll
    for (int i = 0; i < 32; i += 8)
        tile[ty + i][tx] = f32_to_f16_bits(W[(size_t)(kt + ty + i) * N + nt + tx]);
    __syncthreads();
#pragma unroll
    for (int i = 0; i < 32; i += 8)
        Wt[(size_t)(nt + ty + i) * K + kt + tx] = tile[tx][ty + i];
}

// -------------------------------------------------------------------------
// Kernel 3: GEMM [M,K]f16 x Wt[N,K]f16 -> f32, fused bias + exact GELU.
// Block tile 128x64, BK=64, 256 threads = 8 waves.  Ping-pong LDS buffers,
// one barrier per K-tile; per K-step all fragments are register-resident
// before 4 back-to-back v_wmma_f32_16x16x32_f16.
// -------------------------------------------------------------------------
#define BM 128
#define BN 64
#define BK 64
#define LDA 72   // 64 + 8 pad halves -> 144B row stride (16B aligned, conflict-free)
#define LDB 72

__global__ __launch_bounds__(256) void gemm_f16_wmma_kernel(
    const unsigned short* __restrict__ A,   // [M][K] f16
    const unsigned short* __restrict__ Bt,  // [N][K] f16 (pre-transposed W)
    const float* __restrict__ bias,         // [N]
    float* __restrict__ C,                  // [M][N] f32
    int M, int N, int K)
{
    __shared__ unsigned short As[2][BM][LDA];  // 36 KB
    __shared__ unsigned short Bs[2][BN][LDB];  // 18 KB

    const int tid  = threadIdx.x;
    const int wave = tid >> 5;
    const int lane = tid & 31;
    const int bm   = blockIdx.y * BM;
    const int bn   = blockIdx.x * BN;

    // A loader: 2 threads/row, 32 contiguous halves (64B) each
    const int ar = tid >> 1;
    const int ac = (tid & 1) * 32;
    // B loader: 4 threads/row, 16 contiguous halves (32B) each
    const int br = tid >> 2;
    const int bc = (tid & 3) * 16;

    const int arow  = wave * 16 + (lane & 15);
    const int khalf = lane >> 4;            // which K-half this lane holds
    const int coll  = lane & 15;

    // Accumulators pre-loaded with bias (all 8 rows of a tile share lane's col)
    v8f acc[4];
#pragma unroll
    for (int nt = 0; nt < 4; ++nt) {
        const float bz = bias[bn + nt * 16 + coll];
#pragma unroll
        for (int r = 0; r < 8; ++r) acc[nt][r] = bz;
    }

    const unsigned short* gA = A  + (size_t)(bm + ar) * K + ac;
    const unsigned short* gB = Bt + (size_t)(bn + br) * K + bc;

    // ---- Prologue: stage tile 0 into buffer 0 --------------------------
    {
        const uint4* ga = (const uint4*)gA;
        uint4 a0 = ga[0], a1 = ga[1], a2 = ga[2], a3 = ga[3];
        const uint4* gb = (const uint4*)gB;
        uint4 b0 = gb[0], b1 = gb[1];
        *(uint4*)&As[0][ar][ac +  0] = a0;
        *(uint4*)&As[0][ar][ac +  8] = a1;
        *(uint4*)&As[0][ar][ac + 16] = a2;
        *(uint4*)&As[0][ar][ac + 24] = a3;
        *(uint4*)&Bs[0][br][bc +  0] = b0;
        *(uint4*)&Bs[0][br][bc +  8] = b1;
    }

    int buf = 0;
    for (int k0 = 0; k0 < K; k0 += BK, buf ^= 1) {
        const bool has_next = (k0 + BK < K);

        // Next tile -> registers (latency overlaps this tile's WMMAs)
        uint4 na0, na1, na2, na3, nb0, nb1;
        if (has_next) {
            const uint4* ga = (const uint4*)(gA + k0 + BK);
            na0 = ga[0]; na1 = ga[1]; na2 = ga[2]; na3 = ga[3];
            const uint4* gb = (const uint4*)(gB + k0 + BK);
            nb0 = gb[0]; nb1 = gb[1];
        }
        if (k0 + 2 * BK < K) {   // distance-2 prefetch (global_prefetch_b8)
            __builtin_prefetch(gA + k0 + 2 * BK, 0, 1);
            __builtin_prefetch(gB + k0 + 2 * BK, 0, 1);
        }

        __syncthreads();   // stores to As/Bs[buf] complete; [buf^1] reads done

        if (has_next) {    // stage next tile into the other buffer
            *(uint4*)&As[buf ^ 1][ar][ac +  0] = na0;
            *(uint4*)&As[buf ^ 1][ar][ac +  8] = na1;
            *(uint4*)&As[buf ^ 1][ar][ac + 16] = na2;
            *(uint4*)&As[buf ^ 1][ar][ac + 24] = na3;
            *(uint4*)&Bs[buf ^ 1][br][bc +  0] = nb0;
            *(uint4*)&Bs[buf ^ 1][br][bc +  8] = nb1;
        }

#pragma unroll
        for (int s = 0; s < 2; ++s) {        // two 16x16x32 K-steps per tile
            // A fragment (16x32): lane<16 holds K={0..15} of row (lane&15),
            // lane>=16 holds K={16..31}; pairs contiguous in LDS row.
            v16h af;
            {
                const h2v* ap = (const h2v*)&As[buf][arow][0];
                const int base = s * 16 + khalf * 8;   // in h2 (u32) units
#pragma unroll
                for (int j = 0; j < 4; ++j) {
                    h2v p = ap[base + j];
                    h2v q = ap[base + 4 + j];
                    af[2*j + 0] = p[0]; af[2*j + 1] = p[1];
                    af[2*j + 8] = q[0]; af[2*j + 9] = q[1];
                }
            }
            // All 4 B fragments (32x16) into registers first...
            v16h bf[4];
#pragma unroll
            for (int nt = 0; nt < 4; ++nt) {
                const h2v* bp = (const h2v*)&Bs[buf][nt * 16 + coll][0];
                const int base = s * 16 + khalf * 8;
#pragma unroll
                for (int j = 0; j < 8; ++j) {
                    h2v p = bp[base + j];
                    bf[nt][2*j + 0] = p[0];
                    bf[nt][2*j + 1] = p[1];
                }
            }
            // ...then 4 back-to-back WMMAs (single dscnt wait, full XDL issue)
#pragma unroll
            for (int nt = 0; nt < 4; ++nt) {
                acc[nt] = __builtin_amdgcn_wmma_f32_16x16x32_f16(
                    false, af, false, bf[nt], (short)0, acc[nt], false, false);
            }
        }
    }

    // Epilogue: exact GELU (branch-free), f32 store.
    // D layout: VGPR r: lanes 0-15 -> M=r, lanes 16-31 -> M=8+r; N=lane&15.
    const int row0 = bm + wave * 16 + khalf * 8;
#pragma unroll
    for (int nt = 0; nt < 4; ++nt) {
        const int col = bn + nt * 16 + coll;
#pragma unroll
        for (int r = 0; r < 8; ++r) {
            C[(size_t)(row0 + r) * N + col] = gelu_exact(acc[nt][r]);
        }
    }
}

// -------------------------------------------------------------------------
// Launch
// -------------------------------------------------------------------------
extern "C" void kernel_launch(void* const* d_in, const int* in_sizes, int n_in,
                              void* d_out, int out_size, void* d_ws, size_t ws_size,
                              hipStream_t stream) {
    const float* x     = (const float*)d_in[0];
    const float* gamma = (const float*)d_in[1];
    const float* beta  = (const float*)d_in[2];
    const float* W     = (const float*)d_in[3];
    const float* bias  = (const float*)d_in[4];
    float* out = (float*)d_out;

    const int E    = in_sizes[1];          // 512
    const int ntok = in_sizes[0] / E;      // 65536
    const int N    = in_sizes[4];          // 512
    const int K    = 4 * E;                // 2048
    const int M    = ntok / 4;             // 16384

    unsigned short* h  = (unsigned short*)d_ws;                 // M*K f16 (64 MB)
    unsigned short* Wt = h + (size_t)M * K;                     // N*K f16 ( 2 MB)

    ln_to_f16_kernel<<<ntok / 8, 256, 0, stream>>>(x, gamma, beta, h, ntok);

    dim3 tgrid(K / 32, N / 32);            // (64, 16)
    w_transpose_f16_kernel<<<tgrid, 256, 0, stream>>>(W, Wt, K, N);

    dim3 grid(N / BN, M / BM);             // (8, 128) = 1024 blocks
    gemm_f16_wmma_kernel<<<grid, 256, 0, stream>>>(h, Wt, bias, out, M, N, K);
}
